// BitLinear_44074954391797
// MI455X (gfx1250) — 8-bit matrix operand order repaired and checked by reading, compiled, not run on hardware
//
#include <hip/hip_runtime.h>
#include <hip/hip_bf16.h>

typedef int v8i __attribute__((ext_vector_type(8)));

constexpr int BATCH = 16, SEQ = 4096, DIN = 1024, DOUT = 1024;
constexpr int MTOT  = BATCH * SEQ;
constexpr int KBLK  = DIN / 64;

__global__ void quant_swizzle_x(const float* __restrict__ x,
                                const float* __restrict__ act_scale,
                                int* __restrict__ a_swz) {
    const float scale = fmaxf(act_scale[0], 1e-5f);
    const float inv   = 1.0f / scale;

    const int idx = blockIdx.x * blockDim.x + threadIdx.x;
    const int m   = idx >> 8;
    const int k   = (idx & 255) << 2;

    const float4 v = *(const float4*)(x + (size_t)m * DIN + k);
    int q0 = max(-127, min(127, (int)rintf(v.x * inv)));
    int q1 = max(-127, min(127, (int)rintf(v.y * inv)));
    int q2 = max(-127, min(127, (int)rintf(v.z * inv)));
    int q3 = max(-127, min(127, (int)rintf(v.w * inv)));
    unsigned packed = (unsigned)(q0 & 0xff)        | ((unsigned)(q1 & 0xff) << 8) |
                      ((unsigned)(q2 & 0xff) << 16) | ((unsigned)(q3 & 0xff) << 24);

    const int kb = k >> 6, k_in = k & 63;
    const int mtile = m >> 4, m_in = m & 15;
    const int g32 = k_in >> 5, r  = k_in & 31;
    const int g16 = r >> 4,    r2 = r & 15;
    const int h   = r2 >> 3,   r3 = r2 & 7;
    const int vgpr = 4 * g32 + 2 * g16 + (r3 >> 2);
    const int lane = h * 16 + m_in;

    a_swz[(size_t)(mtile * KBLK + kb) * 256 + lane * 8 + vgpr] = (int)packed;
}

__global__ void pack_swizzle_w(const int* __restrict__ tw,
                               int* __restrict__ w_swz) {
    const int idx = blockIdx.x * blockDim.x + threadIdx.x;
    const int o = idx >> 8;
    const int i = (idx & 255) << 2;

    const int4 t = *(const int4*)(tw + (size_t)o * DIN + i);
    unsigned packed = (unsigned)((t.x - 1) & 0xff)        |
                      ((unsigned)((t.y - 1) & 0xff) << 8)  |
                      ((unsigned)((t.z - 1) & 0xff) << 16) |
                      ((unsigned)((t.w - 1) & 0xff) << 24);

    const int kb = i >> 6, k_in = i & 63;
    const int ntile = o >> 4, n = o & 15;
    const int g = k_in >> 5, r  = k_in & 31;
    const int g16 = r >> 4,  r2 = r & 15;
    const int h = r2 >> 3,   r3 = r2 & 7;
    const int vgpr = 4 * g + 2 * g16 + (r3 >> 2);
    const int lane = h * 16 + n;

    w_swz[(size_t)(ntile * KBLK + kb) * 256 + lane * 8 + vgpr] = (int)packed;
}

__global__ void __launch_bounds__(256, 2)
bitlinear_gemm(const int* __restrict__ a_swz,
               const int* __restrict__ w_swz,
               const float* __restrict__ bias,
               const float* __restrict__ act_scale,
               float* __restrict__ out) {
    const int lane   = threadIdx.x & 31;
    const int wave   = threadIdx.x >> 5;
    const int mtileA = (blockIdx.x * 8 + wave) * 2;
    const int mtileB = mtileA + 1;
    const int nt0    = blockIdx.y * 4;

    const float scale = fmaxf(act_scale[0], 1e-5f);

    v8i accA0 = {0,0,0,0,0,0,0,0}, accA1 = {0,0,0,0,0,0,0,0};
    v8i accA2 = {0,0,0,0,0,0,0,0}, accA3 = {0,0,0,0,0,0,0,0};
    v8i accB0 = {0,0,0,0,0,0,0,0}, accB1 = {0,0,0,0,0,0,0,0};
    v8i accB2 = {0,0,0,0,0,0,0,0}, accB3 = {0,0,0,0,0,0,0,0};

    const v8i* __restrict__ Af = (const v8i*)a_swz;
    const v8i* __restrict__ Wf = (const v8i*)w_swz;

    #pragma unroll
    for (int kb = 0; kb < KBLK; ++kb) {
        const v8i aA = Af[(size_t)(mtileA * KBLK + kb) * 32 + lane];
        const v8i aB = Af[(size_t)(mtileB * KBLK + kb) * 32 + lane];
        const v8i b0 = Wf[(size_t)((nt0 + 0) * KBLK + kb) * 32 + lane];
        const v8i b1 = Wf[(size_t)((nt0 + 1) * KBLK + kb) * 32 + lane];
        const v8i b2 = Wf[(size_t)((nt0 + 2) * KBLK + kb) * 32 + lane];
        const v8i b3 = Wf[(size_t)((nt0 + 3) * KBLK + kb) * 32 + lane];
        accA0 = __builtin_amdgcn_wmma_i32_16x16x64_iu8(true, aA, true, b0, accA0, false, false);
        accB0 = __builtin_amdgcn_wmma_i32_16x16x64_iu8(true, aB, true, b0, accB0, false, false);
        accA1 = __builtin_amdgcn_wmma_i32_16x16x64_iu8(true, aA, true, b1, accA1, false, false);
        accB1 = __builtin_amdgcn_wmma_i32_16x16x64_iu8(true, aB, true, b1, accB1, false, false);
        accA2 = __builtin_amdgcn_wmma_i32_16x16x64_iu8(true, aA, true, b2, accA2, false, false);
        accB2 = __builtin_amdgcn_wmma_i32_16x16x64_iu8(true, aB, true, b2, accB2, false, false);
        accA3 = __builtin_amdgcn_wmma_i32_16x16x64_iu8(true, aA, true, b3, accA3, false, false);
        accB3 = __builtin_amdgcn_wmma_i32_16x16x64_iu8(true, aB, true, b3, accB3, false, false);
    }

    const int ncol   = lane & 15;
    const int rbaseA = mtileA * 16 + 8 * (lane >> 4);
    const int rbaseB = mtileB * 16 + 8 * (lane >> 4);
    const float bv0 = bias[(nt0 + 0) * 16 + ncol];
    const float bv1 = bias[(nt0 + 1) * 16 + ncol];
    const float bv2 = bias[(nt0 + 2) * 16 + ncol];
    const float bv3 = bias[(nt0 + 3) * 16 + ncol];

    #pragma unroll
    for (int v = 0; v < 8; ++v) {
        float* oA = out + (size_t)(rbaseA + v) * DOUT;
        oA[(nt0 + 0) * 16 + ncol] = (float)accA0[v] * scale + bv0;
        oA[(nt0 + 1) * 16 + ncol] = (float)accA1[v] * scale + bv1;
        oA[(nt0 + 2) * 16 + ncol] = (float)accA2[v] * scale + bv2;
        oA[(nt0 + 3) * 16 + ncol] = (float)accA3[v] * scale + bv3;
        float* oB = out + (size_t)(rbaseB + v) * DOUT;
        oB[(nt0 + 0) * 16 + ncol] = (float)accB0[v] * scale + bv0;
        oB[(nt0 + 1) * 16 + ncol] = (float)accB1[v] * scale + bv1;
        oB[(nt0 + 2) * 16 + ncol] = (float)accB2[v] * scale + bv2;
        oB[(nt0 + 3) * 16 + ncol] = (float)accB3[v] * scale + bv3;
    }
}

extern "C" void kernel_launch(void* const* d_in, const int* in_sizes, int n_in,
                              void* d_out, int out_size, void* d_ws, size_t ws_size,
                              hipStream_t stream) {
    const float* x         = (const float*)d_in[0];
    const int*   tw        = (const int*)  d_in[1];
    const float* bias      = (const float*)d_in[2];
    const float* act_scale = (const float*)d_in[3];
    float*       out       = (float*)d_out;

    int* a_swz = (int*)d_ws;
    int* w_swz = a_swz + (size_t)MTOT * DIN / 4;

    {
        const int total = MTOT * (DIN / 4);
        quant_swizzle_x<<<total / 256, 256, 0, stream>>>(x, act_scale, a_swz);
    }
    {
        const int total = DOUT * (DIN / 4);
        pack_swizzle_w<<<total / 256, 256, 0, stream>>>(tw, w_swz);
    }
    {
        dim3 grid(MTOT / 32 / 8, DOUT / 64);
        bitlinear_gemm<<<grid, 256, 0, stream>>>(a_swz, w_swz, bias, act_scale, out);
    }
}
